// WindowAttentionLite_10496900071933
// MI455X (gfx1250) — compile-verified, matrix operand order checked by reading, not run on hardware
//
#include <hip/hip_runtime.h>
#include <hip/hip_bf16.h>

// MI455X / gfx1250 wave32 WMMA implementation of windowed attention.
// Pipeline: [K0] weight f32->bf16, [K1] QKV GEMM (bf16 WMMA, double-buffered
// LDS-transposed B), [K2] per-(window,head) attention (WMMA QK^T + softmax +
// WMMA PV), [K3] output projection GEMM (bf16 WMMA, direct-global B fragments).
// Workspace: 512KB weights + 3 * 128MB q/k/v bf16 (~403 MB total).

typedef __attribute__((ext_vector_type(16))) __bf16 v16bf;
typedef __attribute__((ext_vector_type(2)))  __bf16 v2bf;
typedef __attribute__((ext_vector_type(8)))  float  v8f;
typedef unsigned short u16;
typedef unsigned int   u32;

#define NH    8
#define HD    32
#define HW    65536             // 256*256 pixels per image
#define QKVN  (32u*65536u*32u)  // elements per q/k/v buffer (u16)

__device__ __forceinline__ u16 f2bf(float x) {   // fp32 -> bf16 RNE (scalar)
  u32 u = __builtin_bit_cast(u32, x);
  u32 r = u + 0x7FFFu + ((u >> 16) & 1u);
  return (u16)(r >> 16);
}

// Packed conversion: one v_cvt_pk_bf16_f32 per pair (builtin if exposed,
// else direct ISA via inline asm -- CDNA5 has first-class bf16 VALU).
__device__ __forceinline__ u32 f2bf2(float lo, float hi) {
#if __has_builtin(__builtin_amdgcn_cvt_pk_bf16_f32)
  v2bf p = __builtin_amdgcn_cvt_pk_bf16_f32(lo, hi);
  return __builtin_bit_cast(u32, p);
#else
  u32 r;
  asm("v_cvt_pk_bf16_f32 %0, %1, %2" : "=v"(r) : "v"(lo), "v"(hi));
  return r;
#endif
}

union FragBF { v16bf v; uint4 q[2]; };

// A/B fragment: elems 0..7 from p0, elems 8..15 from p0+16 (ISA 16-bit layout,
// lane supplies base = row*stride + 8*(lane>>4); chunks are 16B-aligned b128s).
__device__ __forceinline__ v16bf load_frag16(const u16* p0) {
  FragBF f;
  f.q[0] = *(const uint4*)(p0);
  f.q[1] = *(const uint4*)(p0 + 16);
  return f.v;
}

__device__ __forceinline__ v8f wmma_bf16(v16bf a, v16bf b, v8f c) {
  return __builtin_amdgcn_wmma_f32_16x16x32_bf16(false, a, false, b, (short)0, c,
                                                 false, false);
}

// ---------------- K0: weight conversion ----------------
__global__ void wcvt_kernel(const float* __restrict__ wq, const float* __restrict__ wp,
                            u16* __restrict__ wq_bf, u16* __restrict__ wp_bf) {
  int i = blockIdx.x * blockDim.x + threadIdx.x;
  if (i < 768 * 256) wq_bf[i] = f2bf(wq[i]);
  if (i < 256 * 256) wp_bf[i] = f2bf(wp[i]);
}

// ---------------- K1: QKV projection GEMM ----------------
// out[o, p] = sum_c w[o,c] * x[b,c,p]; block computes 64(M) x 128(N) tile.
// X K-slabs are double-buffered through LDS (transposed to [n][k] bf16) so the
// next slab's global_load_b128s overlap the current slab's WMMAs.
// Result stored bf16 as qkv[sel][(b*8+h)*HW + p][d] so attention/proj fragment
// loads are contiguous 16B chunks along d.
__launch_bounds__(256)
__global__ void qkv_kernel(const float* __restrict__ x, const u16* __restrict__ wbf,
                           u16* __restrict__ qkv_ws) {
  __shared__ u16 Xt[2][128 * 32];            // double-buffered B tile [n][k]
  const int b     = blockIdx.z;
  const int mblk  = blockIdx.y;              // 12 tiles of 64 over M=768
  const int pbase = blockIdx.x * 128;
  const int lane  = threadIdx.x & 31;
  const int wid   = threadIdx.x >> 5;
  const int half  = lane >> 4;
  const int l16   = lane & 15;
  const int m_sub  = wid >> 1;               // wave's 16-row M subtile (0..3)
  const int ntile0 = (wid & 1) * 4;          // wave's 4 N subtiles

  // Each thread stages 4 float4 chunks per slab: chunk q -> (k = q>>5, n0 = (q&31)*4)
  const int sk[4] = { (int)threadIdx.x >> 5, ((int)threadIdx.x + 256) >> 5,
                      ((int)threadIdx.x + 512) >> 5, ((int)threadIdx.x + 768) >> 5 };
  const int sn = (threadIdx.x & 31) * 4;

  v8f acc[4];
  const v8f vzero = {0.f,0.f,0.f,0.f,0.f,0.f,0.f,0.f};
#pragma unroll
  for (int nt = 0; nt < 4; ++nt) acc[nt] = vzero;

  const float* xb = x + ((size_t)(b * 256) * HW) + pbase;

  // Prologue: stage slab 0.
  {
    float4 ld[4];
#pragma unroll
    for (int ii = 0; ii < 4; ++ii)
      ld[ii] = *(const float4*)(xb + (size_t)sk[ii] * HW + sn);
#pragma unroll
    for (int ii = 0; ii < 4; ++ii) {
      u32 p01 = f2bf2(ld[ii].x, ld[ii].y), p23 = f2bf2(ld[ii].z, ld[ii].w);
      Xt[0][(sn + 0) * 32 + sk[ii]] = (u16)p01;
      Xt[0][(sn + 1) * 32 + sk[ii]] = (u16)(p01 >> 16);
      Xt[0][(sn + 2) * 32 + sk[ii]] = (u16)p23;
      Xt[0][(sn + 3) * 32 + sk[ii]] = (u16)(p23 >> 16);
    }
  }
  __syncthreads();

  for (int kb = 0; kb < 8; ++kb) {
    // Issue next slab's global loads before touching the current buffer.
    float4 ld[4];
    if (kb < 7) {
      const float* xsrc = xb + (size_t)((kb + 1) * 32) * HW;
#pragma unroll
      for (int ii = 0; ii < 4; ++ii)
        ld[ii] = *(const float4*)(xsrc + (size_t)sk[ii] * HW + sn);
    }

    const int orow = mblk * 64 + m_sub * 16 + l16;
    v16bf afrag = load_frag16(wbf + (size_t)orow * 256 + kb * 32 + 8 * half);
    const u16* xt = Xt[kb & 1];
#pragma unroll
    for (int nt = 0; nt < 4; ++nt) {
      const u16* bp = &xt[((ntile0 + nt) * 16 + l16) * 32 + 8 * half];
      acc[nt] = wmma_bf16(afrag, load_frag16(bp), acc[nt]);
    }

    if (kb < 7) {
      u16* xn = Xt[(kb + 1) & 1];
#pragma unroll
      for (int ii = 0; ii < 4; ++ii) {
        u32 p01 = f2bf2(ld[ii].x, ld[ii].y), p23 = f2bf2(ld[ii].z, ld[ii].w);
        xn[(sn + 0) * 32 + sk[ii]] = (u16)p01;
        xn[(sn + 1) * 32 + sk[ii]] = (u16)(p01 >> 16);
        xn[(sn + 2) * 32 + sk[ii]] = (u16)p23;
        xn[(sn + 3) * 32 + sk[ii]] = (u16)(p23 >> 16);
      }
    }
    __syncthreads();
  }

  // D layout: lane holds 8 consecutive o (=d) values for fixed pixel p.
  const int o0  = mblk * 64 + m_sub * 16 + 8 * half;
  const int sel = o0 >> 8;                   // 0=q 1=k 2=v
  const int h   = (o0 & 255) >> 5;
  const int d0  = o0 & 31;
#pragma unroll
  for (int nt = 0; nt < 4; ++nt) {
    int p = pbase + (ntile0 + nt) * 16 + l16;
    uint4 dw;
    dw.x = f2bf2(acc[nt][0], acc[nt][1]);
    dw.y = f2bf2(acc[nt][2], acc[nt][3]);
    dw.z = f2bf2(acc[nt][4], acc[nt][5]);
    dw.w = f2bf2(acc[nt][6], acc[nt][7]);
    size_t dst = (size_t)sel * QKVN + (((size_t)(b * 8 + h)) * HW + p) * 32 + d0;
    *(uint4*)(qkv_ws + dst) = dw;
  }
}

// ---------------- K2: windowed attention ----------------
// One wave = one (window, head): S=QK^T (16 WMMA), softmax, O=PV (32 WMMA).
// O is bounced through LDS (reusing consumed P rows) and written back over the
// Q buffer as dense b128 stores (128B contiguous per lane).
__launch_bounds__(128)
__global__ void attn_kernel(u16* __restrict__ q_ws, const u16* __restrict__ k_ws,
                            const u16* __restrict__ v_ws) {
  __shared__ u16 sm[4 * 6144];               // per wave: P[64*64] + Vt[32*64]
  const int lane = threadIdx.x & 31;
  const int wid  = threadIdx.x >> 5;
  const int half = lane >> 4;
  const int l16  = lane & 15;
  const int win  = blockIdx.x;               // 4096 windows
  const int b    = win >> 10;
  const int wh   = (win >> 5) & 31;
  const int ww   = win & 31;
  const int head = blockIdx.y * 4 + wid;
  const size_t hb = (size_t)(b * 8 + head) * HW;
  u16* P  = sm + wid * 6144;
  u16* Vt = P + 4096;
  const float scale = 0.17677669529663687f;  // 1/sqrt(32)
  const int prow0 = wh * 8 * 256 + ww * 8;
  auto PIX = [&](int n) { return prow0 + (n >> 3) * 256 + (n & 7); };
  const v8f vzero = {0.f,0.f,0.f,0.f,0.f,0.f,0.f,0.f};

  // K as B-matrix fragments (lane = token, contiguous d chunks)
  v16bf kf[4];
#pragma unroll
  for (int j = 0; j < 4; ++j)
    kf[j] = load_frag16(k_ws + (hb + PIX(j * 16 + l16)) * 32 + 8 * half);

  // Stage V transposed into LDS: Vt[d][token]. Each lane owns two adjacent
  // tokens and packs them into one dword per d -> 32 aligned ds_store_b32s.
  {
    const int t0 = lane * 2;
    const u16* vpA = v_ws + (hb + PIX(t0)) * 32;
    const u16* vpB = v_ws + (hb + PIX(t0 + 1)) * 32;
    alignas(16) u32 wA[16], wB[16];
    *(uint4*)&wA[0]  = *(const uint4*)(vpA);
    *(uint4*)&wA[4]  = *(const uint4*)(vpA + 8);
    *(uint4*)&wA[8]  = *(const uint4*)(vpA + 16);
    *(uint4*)&wA[12] = *(const uint4*)(vpA + 24);
    *(uint4*)&wB[0]  = *(const uint4*)(vpB);
    *(uint4*)&wB[4]  = *(const uint4*)(vpB + 8);
    *(uint4*)&wB[8]  = *(const uint4*)(vpB + 16);
    *(uint4*)&wB[12] = *(const uint4*)(vpB + 24);
#pragma unroll
    for (int j = 0; j < 16; ++j) {
      u32 lo2 = (wA[j] & 0xffffu) | (wB[j] << 16);       // d = 2j
      u32 hi2 = (wA[j] >> 16) | (wB[j] & 0xffff0000u);   // d = 2j+1
      *(u32*)(Vt + (2 * j + 0) * 64 + t0) = lo2;
      *(u32*)(Vt + (2 * j + 1) * 64 + t0) = hi2;
    }
  }

  // S = QK^T, row softmax, P -> LDS (bf16, re-shaped D-layout -> A-layout)
#pragma unroll
  for (int i = 0; i < 4; ++i) {
    v16bf qf = load_frag16(q_ws + (hb + PIX(i * 16 + l16)) * 32 + 8 * half);
    v8f s[4];
#pragma unroll
    for (int j = 0; j < 4; ++j) s[j] = wmma_bf16(qf, kf[j], vzero);
#pragma unroll
    for (int r = 0; r < 8; ++r) {
      float s0 = s[0][r] * scale, s1 = s[1][r] * scale;
      float s2 = s[2][r] * scale, s3 = s[3][r] * scale;
      float mx = fmaxf(fmaxf(s0, s1), fmaxf(s2, s3));
      mx = fmaxf(mx, __shfl_xor(mx, 1));   // masks 1..8 stay within the
      mx = fmaxf(mx, __shfl_xor(mx, 2));   // 16-lane half = one matrix row
      mx = fmaxf(mx, __shfl_xor(mx, 4));
      mx = fmaxf(mx, __shfl_xor(mx, 8));
      float e0 = __expf(s0 - mx), e1 = __expf(s1 - mx);
      float e2 = __expf(s2 - mx), e3 = __expf(s3 - mx);
      float sum = e0 + e1 + e2 + e3;
      sum += __shfl_xor(sum, 1);
      sum += __shfl_xor(sum, 2);
      sum += __shfl_xor(sum, 4);
      sum += __shfl_xor(sum, 8);
      float rs = 1.0f / sum;
      u32 p01 = f2bf2(e0 * rs, e1 * rs);
      u32 p23 = f2bf2(e2 * rs, e3 * rs);
      int mrow = i * 16 + 8 * half + r;
      P[mrow * 64 +  0 + l16] = (u16)p01;
      P[mrow * 64 + 16 + l16] = (u16)(p01 >> 16);
      P[mrow * 64 + 32 + l16] = (u16)p23;
      P[mrow * 64 + 48 + l16] = (u16)(p23 >> 16);
    }
  }

  // V B-fragments from LDS (ds_load_b128 pairs)
  v16bf vf[4][2];
#pragma unroll
  for (int kt = 0; kt < 4; ++kt)
#pragma unroll
    for (int nt = 0; nt < 2; ++nt)
      vf[kt][nt] = load_frag16(Vt + (nt * 16 + l16) * 64 + kt * 32 + 8 * half);

  // O = P * V. Stage O rows into Obuf (=P region; row-block i only overwrites
  // P rows < 16(i+1)/2, all consumed by earlier A-fragment reads).
  u16* Obuf = P;                             // [64 tokens][32 d]
#pragma unroll
  for (int i = 0; i < 4; ++i) {
    v8f o0 = vzero, o1 = vzero;
#pragma unroll
    for (int kt = 0; kt < 4; ++kt) {
      v16bf pf = load_frag16(P + (i * 16 + l16) * 64 + kt * 32 + 8 * half);
      o0 = wmma_bf16(pf, vf[kt][0], o0);
      o1 = wmma_bf16(pf, vf[kt][1], o1);
    }
    int tok0 = i * 16 + 8 * half;
#pragma unroll
    for (int r = 0; r < 8; ++r) {
      u32 p01 = f2bf2(o0[r], o1[r]);
      Obuf[(tok0 + r) * 32 + l16]      = (u16)p01;         // d = l16
      Obuf[(tok0 + r) * 32 + 16 + l16] = (u16)(p01 >> 16); // d = 16 + l16
    }
  }

  // Coalesced writeback over the Q buffer: each lane owns two adjacent tokens
  // -> 128B contiguous global_store_b128 traffic.
#pragma unroll
  for (int t = 0; t < 2; ++t) {
    int tok = lane * 2 + t;
    const u16* src = Obuf + tok * 32;
    u16* dst = q_ws + (hb + PIX(tok)) * 32;
    uint4 c0 = *(const uint4*)(src);
    uint4 c1 = *(const uint4*)(src + 8);
    uint4 c2 = *(const uint4*)(src + 16);
    uint4 c3 = *(const uint4*)(src + 24);
    *(uint4*)(dst)      = c0;
    *(uint4*)(dst + 8)  = c1;
    *(uint4*)(dst + 16) = c2;
    *(uint4*)(dst + 24) = c3;
  }
}

// ---------------- K3: output projection GEMM ----------------
// K-tiles of 32 align with heads -> B fragments come straight from global.
__launch_bounds__(256)
__global__ void proj_kernel(const u16* __restrict__ attn, const u16* __restrict__ wpbf,
                            float* __restrict__ out) {
  const int b     = blockIdx.z;
  const int mblk  = blockIdx.y;              // 4 tiles of 64 over M=256
  const int pbase = blockIdx.x * 128;
  const int lane  = threadIdx.x & 31;
  const int wid   = threadIdx.x >> 5;
  const int half  = lane >> 4;
  const int l16   = lane & 15;
  const int m_sub  = wid >> 1;
  const int ntile0 = (wid & 1) * 4;

  v8f acc[4];
  const v8f vzero = {0.f,0.f,0.f,0.f,0.f,0.f,0.f,0.f};
#pragma unroll
  for (int nt = 0; nt < 4; ++nt) acc[nt] = vzero;

#pragma unroll
  for (int kb = 0; kb < 8; ++kb) {           // kb == head index
    const int orow = mblk * 64 + m_sub * 16 + l16;
    v16bf af = load_frag16(wpbf + (size_t)orow * 256 + kb * 32 + 8 * half);
#pragma unroll
    for (int nt = 0; nt < 4; ++nt) {
      int p = pbase + (ntile0 + nt) * 16 + l16;
      const u16* bp = attn + (((size_t)(b * 8 + kb)) * HW + p) * 32 + 8 * half;
      acc[nt] = wmma_bf16(af, load_frag16(bp), acc[nt]);
    }
  }

  const int o0 = mblk * 64 + m_sub * 16 + 8 * half;
#pragma unroll
  for (int nt = 0; nt < 4; ++nt) {
    int p = pbase + (ntile0 + nt) * 16 + l16;
#pragma unroll
    for (int r = 0; r < 8; ++r)
      out[((size_t)(b * 256 + o0 + r)) * HW + p] = acc[nt][r];
  }
}

extern "C" void kernel_launch(void* const* d_in, const int* in_sizes, int n_in,
                              void* d_out, int out_size, void* d_ws, size_t ws_size,
                              hipStream_t stream) {
  (void)in_sizes; (void)n_in; (void)out_size; (void)ws_size;
  const float* x      = (const float*)d_in[0];
  const float* w_qkv  = (const float*)d_in[1];
  const float* w_proj = (const float*)d_in[2];
  float* out = (float*)d_out;

  char* ws = (char*)d_ws;
  u16* wq_bf = (u16*)ws;                      // 768*256 bf16
  u16* wp_bf = wq_bf + 768 * 256;             // 256*256 bf16
  u16* qkv   = (u16*)(ws + 524288);           // q,k,v each 32*HW*32 bf16
  u16* qbuf  = qkv;
  u16* kbuf  = qkv + QKVN;
  u16* vbuf  = qkv + 2 * (size_t)QKVN;

  wcvt_kernel<<<768, 256, 0, stream>>>(w_qkv, w_proj, wq_bf, wp_bf);
  qkv_kernel<<<dim3(512, 12, 4), 256, 0, stream>>>(x, wq_bf, qkv);
  attn_kernel<<<dim3(4096, 2), 128, 0, stream>>>(qbuf, kbuf, vbuf);
  proj_kernel<<<dim3(512, 4, 4), 256, 0, stream>>>(qbuf, wp_bf, out);
}
